// UnetMamba_68702296867364
// MI455X (gfx1250) — matrix operand order repaired and checked by reading, not compiled, not run on hardware
//
#include <hip/hip_runtime.h>
#include <cstddef>
#include <cstdint>

// ---------------------------------------------------------------------------
// CDNA5 / gfx1250 implementation. Heavy GEMMs (spatial MLPs, attention
// qkv/proj, head linears) use v_wmma_f32_16x16x32_bf16 with f32 accumulate.
// Wave computes a 16x64 strip (4 WMMA tiles) sharing one A fragment; the
// K-loop is manually unrolled x2 with ping-pong register buffers so global
// loads overlap the WMMAs with no rotation moves. Requires K % 64 == 0.
// ---------------------------------------------------------------------------

typedef __attribute__((ext_vector_type(16))) __bf16 v16bf;
typedef __attribute__((ext_vector_type(8)))  float  v8f;
typedef __attribute__((ext_vector_type(4)))  float  v4f;

__device__ __forceinline__ float gelu_erf(float x) {
  return 0.5f * x * (1.0f + erff(x * 0.7071067811865475f));
}

// -------------------------- f32 -> bf16 conversion -------------------------
__global__ void k_f32_to_bf16(const float* __restrict__ x, __bf16* __restrict__ y, size_t n) {
  size_t i = (size_t)blockIdx.x * blockDim.x + threadIdx.x;
  if (i < n) y[i] = (__bf16)x[i];
}

// -------------------------- WMMA GEMM --------------------------------------
// C[M,N] = post( A[M,K](f32) x W[N,K]^T(bf16) + bias + res ),  K % 64 == 0.
// Block: 256 thr = 8 waves, arranged 4(M) x 2(N): block tile 64M x 128N.
__global__ void __launch_bounds__(256)
k_gemm_wmma(const float* __restrict__ A, const __bf16* __restrict__ W,
            const float* __restrict__ bias, const float* __restrict__ res,
            float* __restrict__ C, int M, int N, int K, int postop) {
  const int lane = threadIdx.x & 31;
  const int wave = threadIdx.x >> 5;
  const int m0 = blockIdx.y * 64 + (wave >> 1) * 16;
  const int n0 = blockIdx.x * 128 + (wave & 1) * 64;
  const int half = lane >> 4;   // 0 or 1
  const int l16  = lane & 15;

  int ar = m0 + l16; if (ar > M - 1) ar = M - 1;
  const float* Ap = A + (size_t)ar * K + half * 16;   // A frag: K = half*16 + [0..15]

  const __bf16* Wp[4];
#pragma unroll
  for (int t = 0; t < 4; ++t) {
    int wr = n0 + t * 16 + l16; if (wr > N - 1) wr = N - 1; if (wr < 0) wr = 0;
    Wp[t] = W + (size_t)wr * K + half * 16;           // B frag: K = half*16 + [0..15]
  }

  const v8f vz = {0.f, 0.f, 0.f, 0.f, 0.f, 0.f, 0.f, 0.f};
  v8f acc[4];
#pragma unroll
  for (int t = 0; t < 4; ++t) acc[t] = vz;

  // ---- prologue: fill buffer X with K-step 0 ----
  v4f a0 = *(const v4f*)(Ap + 0);
  v4f a1 = *(const v4f*)(Ap + 4);
  v4f a2 = *(const v4f*)(Ap + 8);
  v4f a3 = *(const v4f*)(Ap + 12);
  v16bf bx[4];
#pragma unroll
  for (int t = 0; t < 4; ++t) bx[t] = *(const v16bf*)(Wp[t]);

  for (int k = 0; k < K; k += 64) {
    // ---- issue buffer Y loads (K-step k+32; always in range, K%64==0) ----
    const int k1 = k + 32;
    v4f c0 = *(const v4f*)(Ap + k1);
    v4f c1 = *(const v4f*)(Ap + k1 + 4);
    v4f c2 = *(const v4f*)(Ap + k1 + 8);
    v4f c3 = *(const v4f*)(Ap + k1 + 12);
    v16bf by[4];
#pragma unroll
    for (int t = 0; t < 4; ++t) by[t] = *(const v16bf*)(Wp[t] + k1);
    __builtin_prefetch(Ap + k1 + 32, 0, 1);

    // ---- compute with buffer X ----
    v16bf af;
#pragma unroll
    for (int j = 0; j < 4; ++j) {
      af[j]      = (__bf16)a0[j];
      af[4 + j]  = (__bf16)a1[j];
      af[8 + j]  = (__bf16)a2[j];
      af[12 + j] = (__bf16)a3[j];
    }
#pragma unroll
    for (int t = 0; t < 4; ++t)
      acc[t] = __builtin_amdgcn_wmma_f32_16x16x32_bf16(false, af, false, bx[t],
                                                       (short)0, acc[t], false, false);

    // ---- issue buffer X loads (K-step k+64; clamp on last iteration) ----
    const int k2 = (k + 64 < K) ? (k + 64) : k;
    a0 = *(const v4f*)(Ap + k2);
    a1 = *(const v4f*)(Ap + k2 + 4);
    a2 = *(const v4f*)(Ap + k2 + 8);
    a3 = *(const v4f*)(Ap + k2 + 12);
#pragma unroll
    for (int t = 0; t < 4; ++t) bx[t] = *(const v16bf*)(Wp[t] + k2);
    __builtin_prefetch(Ap + k2 + 32, 0, 1);

    // ---- compute with buffer Y ----
    v16bf ag;
#pragma unroll
    for (int j = 0; j < 4; ++j) {
      ag[j]      = (__bf16)c0[j];
      ag[4 + j]  = (__bf16)c1[j];
      ag[8 + j]  = (__bf16)c2[j];
      ag[12 + j] = (__bf16)c3[j];
    }
#pragma unroll
    for (int t = 0; t < 4; ++t)
      acc[t] = __builtin_amdgcn_wmma_f32_16x16x32_bf16(false, ag, false, by[t],
                                                       (short)0, acc[t], false, false);
  }

#pragma unroll
  for (int t = 0; t < 4; ++t) {
    int col = n0 + t * 16 + l16;
    if (col < N) {
      float bv = bias ? bias[col] : 0.f;
#pragma unroll
      for (int i = 0; i < 8; ++i) {
        int row = m0 + i + half * 8;
        if (row < M) {
          float v = acc[t][i] + bv;
          if (res) v += res[(size_t)row * N + col];
          if (postop == 1) v = gelu_erf(v);
          C[(size_t)row * N + col] = v;
        }
      }
    }
  }
}

// -------------------------- Row LayerNorm (+postop) ------------------------
// One 256-thread block per row. postop: 0=none 1=gelu 2=relu. In-place safe.
__global__ void k_ln_rows(const float* __restrict__ x, const float* __restrict__ g,
                          const float* __restrict__ bta, float* __restrict__ y,
                          int E, int postop) {
  __shared__ float red[256];
  const float* xr = x + (size_t)blockIdx.x * E;
  float s = 0.f;
  for (int i = threadIdx.x; i < E; i += 256) s += xr[i];
  red[threadIdx.x] = s; __syncthreads();
  for (int o = 128; o > 0; o >>= 1) { if (threadIdx.x < o) red[threadIdx.x] += red[threadIdx.x + o]; __syncthreads(); }
  float mean = red[0] / E; __syncthreads();
  float v = 0.f;
  for (int i = threadIdx.x; i < E; i += 256) { float d = xr[i] - mean; v += d * d; }
  red[threadIdx.x] = v; __syncthreads();
  for (int o = 128; o > 0; o >>= 1) { if (threadIdx.x < o) red[threadIdx.x] += red[threadIdx.x + o]; __syncthreads(); }
  float rstd = rsqrtf(red[0] / E + 1e-5f);
  float* yr = y + (size_t)blockIdx.x * E;
  for (int i = threadIdx.x; i < E; i += 256) {
    float val = (xr[i] - mean) * rstd * g[i] + bta[i];
    if (postop == 1) val = gelu_erf(val);
    else if (postop == 2) val = fmaxf(val, 0.f);
    yr[i] = val;
  }
}

// ------------------ spatial: phase projection + LN + relu ------------------
// x[B,2,T,E] -> h[B,4,T,E]; one block per (b,o,t), E==256==blockDim.
__global__ void k_phase_ln_relu(const float* __restrict__ x, const float* __restrict__ pw,
                                const float* __restrict__ pb, const float* __restrict__ g,
                                const float* __restrict__ bta, float* __restrict__ h,
                                int B, int T, int E) {
  __shared__ float red[256];
  int idx = blockIdx.x;
  int t = idx % T; int o = (idx / T) % 4; int b = idx / (T * 4);
  const float* x0 = x + (((size_t)b * 2 + 0) * T + t) * E;
  const float* x1 = x + (((size_t)b * 2 + 1) * T + t) * E;
  int i = threadIdx.x;
  float v = x0[i] * pw[o * 2] + x1[i] * pw[o * 2 + 1] + pb[o];
  red[i] = v; __syncthreads();
  for (int off = 128; off > 0; off >>= 1) { if (i < off) red[i] += red[i + off]; __syncthreads(); }
  float mean = red[0] / E; __syncthreads();
  float d = v - mean; red[i] = d * d; __syncthreads();
  for (int off = 128; off > 0; off >>= 1) { if (i < off) red[i] += red[i + off]; __syncthreads(); }
  float rstd = rsqrtf(red[0] / E + 1e-5f);
  float val = (v - mean) * rstd * g[i] + bta[i];
  h[(((size_t)b * 4 + o) * T + t) * (size_t)E + i] = fmaxf(val, 0.f);
}

// ------------------ spatial: mean/max over 4 channels ----------------------
__global__ void k_mean_max_c4(const float* __restrict__ n1, float* __restrict__ cat,
                              int B, int TE) {
  size_t i = (size_t)blockIdx.x * blockDim.x + threadIdx.x;
  if (i >= (size_t)B * TE) return;
  size_t b = i / TE, r = i % TE;
  const float* p = n1 + b * 4 * (size_t)TE + r;
  float a0 = p[0], a1 = p[TE], a2 = p[2 * (size_t)TE], a3 = p[3 * (size_t)TE];
  cat[(b * 2) * (size_t)TE + r]     = 0.25f * (a0 + a1 + a2 + a3);
  cat[(b * 2 + 1) * (size_t)TE + r] = fmaxf(fmaxf(a0, a1), fmaxf(a2, a3));
}

// ------------------ spatial: 7x7 conv gate, h += (tanh(s)+1)*n1 ------------
__global__ void k_spa_gate(const float* __restrict__ n1, const float* __restrict__ cat,
                           const float* __restrict__ w, const float* __restrict__ spab,
                           float* __restrict__ h, int B, int T, int E) {
  size_t i = (size_t)blockIdx.x * blockDim.x + threadIdx.x;
  if (i >= (size_t)B * T * E) return;
  int e = (int)(i % E); int t = (int)((i / E) % T); int b = (int)(i / ((size_t)E * T));
  float s = 0.f;
  for (int c = 0; c < 2; ++c)
    for (int dy = 0; dy < 7; ++dy) {
      int tt = t + dy - 3; if (tt < 0 || tt >= T) continue;
      const float* cr = cat + (((size_t)b * 2 + c) * T + tt) * E;
      const float* wr = w + (c * 7 + dy) * 7;
      for (int dx = 0; dx < 7; ++dx) {
        int ee = e + dx - 3; if (ee < 0 || ee >= E) continue;
        s += cr[ee] * wr[dx];
      }
    }
  float gfac = tanhf(s + spab[0]) + 1.0f;
  for (int c = 0; c < 4; ++c) {
    size_t o = (((size_t)b * 4 + c) * T + t) * (size_t)E + e;
    h[o] += gfac * n1[o];
  }
}

// ------------------ spatial: mean over 4 channels --------------------------
__global__ void k_mean_c4(const float* __restrict__ x, float* __restrict__ y, int B, int TE) {
  size_t i = (size_t)blockIdx.x * blockDim.x + threadIdx.x;
  if (i >= (size_t)B * TE) return;
  size_t b = i / TE, r = i % TE;
  const float* p = x + b * 4 * (size_t)TE + r;
  y[i] = 0.25f * (p[0] + p[TE] + p[2 * (size_t)TE] + p[3 * (size_t)TE]);
}

// ------------------ concat x[:,0], x[:,1] along length ---------------------
__global__ void k_inc_concat(const float* __restrict__ x, float* __restrict__ y,
                             int B, int C, int E) {
  size_t i = (size_t)blockIdx.x * blockDim.x + threadIdx.x;
  if (i >= (size_t)B * C * 2 * E) return;
  int l = (int)(i % (2 * E)); int c = (int)((i / (2 * E)) % C);
  int b = (int)(i / ((size_t)2 * E * C));
  int part = l / E, e = l % E;
  y[i] = x[(((size_t)b * 2 + part) * C + c) * (size_t)E + e];
}

// ------------------ conv1d k=3 pad=1 (no bias) -----------------------------
__global__ void k_conv1d_k3(const float* __restrict__ x, const float* __restrict__ w,
                            float* __restrict__ y, int B, int Ci, int Co, int L) {
  size_t i = (size_t)blockIdx.x * blockDim.x + threadIdx.x;
  if (i >= (size_t)B * Co * L) return;
  int l = (int)(i % L); int co = (int)((i / L) % Co); int b = (int)(i / ((size_t)L * Co));
  const float* xb = x + (size_t)b * Ci * L;
  const float* wr = w + (size_t)co * Ci * 3;
  float s = 0.f;
  for (int ci = 0; ci < Ci; ++ci) {
    const float* xr = xb + (size_t)ci * L + l;
    float w0 = wr[ci * 3], w1 = wr[ci * 3 + 1], w2 = wr[ci * 3 + 2];
    if (l > 0) s += xr[-1] * w0;
    s += xr[0] * w1;
    if (l < L - 1) s += xr[1] * w2;
  }
  y[i] = s;
}

// ------------------ MaxPool1d(2) over contiguous last axis -----------------
__global__ void k_pool2(const float* __restrict__ x, float* __restrict__ y, size_t n) {
  size_t i = (size_t)blockIdx.x * blockDim.x + threadIdx.x;
  if (i < n) y[i] = fmaxf(x[2 * i], x[2 * i + 1]);
}

// ------------------ attention core (per b,head), templated head_dim --------
template <int HD>
__global__ void k_attn_core(const float* __restrict__ qkv, float* __restrict__ o,
                            int Ntok, int C) {
  int b = blockIdx.x >> 3;      // H == 8
  int h = blockIdx.x & 7;
  float scale = rsqrtf((float)HD);
  for (int i = threadIdx.x; i < Ntok; i += blockDim.x) {
    const float* qr = qkv + ((size_t)(b * Ntok + i) * 3 * C) + h * HD;
    float qv[HD];
#pragma unroll
    for (int d = 0; d < HD; ++d) qv[d] = qr[d] * scale;
    float mx = -1e30f;
    for (int j = 0; j < Ntok; ++j) {
      const float* kr = qkv + ((size_t)(b * Ntok + j) * 3 * C) + C + h * HD;
      float s = 0.f;
#pragma unroll
      for (int d = 0; d < HD; ++d) s += qv[d] * kr[d];
      mx = fmaxf(mx, s);
    }
    float acc[HD];
#pragma unroll
    for (int d = 0; d < HD; ++d) acc[d] = 0.f;
    float l = 0.f;
    for (int j = 0; j < Ntok; ++j) {
      const float* kr = qkv + ((size_t)(b * Ntok + j) * 3 * C) + C + h * HD;
      const float* vr = kr + C;
      float s = 0.f;
#pragma unroll
      for (int d = 0; d < HD; ++d) s += qv[d] * kr[d];
      float e = expf(s - mx);
      l += e;
#pragma unroll
      for (int d = 0; d < HD; ++d) acc[d] += e * vr[d];
    }
    float inv = 1.f / l;
    float* orow = o + (size_t)(b * Ntok + i) * C + h * HD;
#pragma unroll
    for (int d = 0; d < HD; ++d) orow[d] = acc[d] * inv;
  }
}

// ------------------ ConvTranspose1d k=2 s=2 --------------------------------
__global__ void k_convt2(const float* __restrict__ x, const float* __restrict__ w,
                         const float* __restrict__ bias, float* __restrict__ y,
                         int B, int I, int O, int L) {
  size_t i = (size_t)blockIdx.x * blockDim.x + threadIdx.x;
  if (i >= (size_t)B * O * 2 * L) return;
  int m = (int)(i % (2 * L)); int o = (int)((i / (2 * L)) % O);
  int b = (int)(i / ((size_t)2 * L * O));
  int l = m >> 1, j = m & 1;
  float s = bias[o];
  const float* xb = x + (size_t)b * I * L + l;
  for (int ii = 0; ii < I; ++ii) s += xb[(size_t)ii * L] * w[((size_t)ii * O + o) * 2 + j];
  y[i] = s;
}

// ------------------ concat along channel axis ------------------------------
__global__ void k_concat_ch(const float* __restrict__ a, const float* __restrict__ b2,
                            float* __restrict__ y, int B, int C1, int C2, int L) {
  size_t i = (size_t)blockIdx.x * blockDim.x + threadIdx.x;
  int Ct = C1 + C2;
  if (i >= (size_t)B * Ct * L) return;
  int l = (int)(i % L); int c = (int)((i / L) % Ct); int b = (int)(i / ((size_t)L * Ct));
  y[i] = (c < C1) ? a[((size_t)b * C1 + c) * L + l]
                  : b2[((size_t)b * C2 + (c - C1)) * L + l];
}

// ------------------ final conv k=2 s=2 + bias + x_spa ----------------------
__global__ void k_conv_s2_add(const float* __restrict__ x, const float* __restrict__ w,
                              const float* __restrict__ bias, const float* __restrict__ spa,
                              float* __restrict__ y, int B, int Ci, int Co, int Lin) {
  int Lo = Lin / 2;
  size_t i = (size_t)blockIdx.x * blockDim.x + threadIdx.x;
  if (i >= (size_t)B * Co * Lo) return;
  int l = (int)(i % Lo); int o = (int)((i / Lo) % Co); int b = (int)(i / ((size_t)Lo * Co));
  float s = bias[o];
  const float* xb = x + (size_t)b * Ci * Lin + 2 * l;
  const float* wo = w + (size_t)o * Ci * 2;
  for (int ci = 0; ci < Ci; ++ci)
    s += xb[(size_t)ci * Lin] * wo[ci * 2] + xb[(size_t)ci * Lin + 1] * wo[ci * 2 + 1];
  y[i] = s + spa[i];
}

// ------------------ channel attention gate (in-place) ----------------------
__global__ void k_chan_att(float* __restrict__ z, const float* __restrict__ fc1,
                           const float* __restrict__ fc2, int B, int Cc, int E) {
  __shared__ float sm[32], sx[32], hm[5], hx[5], gt[32];
  int b = blockIdx.x;
  int t = threadIdx.x;
  if (t < Cc) {
    const float* zr = z + ((size_t)b * Cc + t) * E;
    float s = 0.f, m = -1e30f;
    for (int e = 0; e < E; ++e) { float v = zr[e]; s += v; m = fmaxf(m, v); }
    sm[t] = s / E; sx[t] = m;
  }
  __syncthreads();
  if (t < 5) {
    float a = 0.f, c2 = 0.f;
    for (int c = 0; c < Cc; ++c) { a += fc1[t * Cc + c] * sm[c]; c2 += fc1[t * Cc + c] * sx[c]; }
    hm[t] = fmaxf(a, 0.f); hx[t] = fmaxf(c2, 0.f);
  }
  __syncthreads();
  if (t < Cc) {
    float s = 0.f;
    for (int k = 0; k < 5; ++k) s += fc2[t * 5 + k] * (hm[k] + hx[k]);
    gt[t] = 1.f / (1.f + expf(-s));
  }
  __syncthreads();
  for (int i = t; i < Cc * E; i += blockDim.x)
    z[(size_t)b * Cc * E + i] *= gt[i / E];
}

// ------------------ final tiny linear (N=4) --------------------------------
__global__ void k_lin3(const float* __restrict__ x, const float* __restrict__ w,
                       const float* __restrict__ bias, float* __restrict__ out,
                       int B, int K, int O) {
  int i = blockIdx.x * blockDim.x + threadIdx.x;
  if (i >= B * O) return;
  int o = i % O, b = i / O;
  float s = bias[o];
  const float* xr = x + (size_t)b * K;
  const float* wr = w + (size_t)o * K;
  for (int k = 0; k < K; ++k) s += xr[k] * wr[k];
  out[i] = s;
}

// ===========================================================================
extern "C" void kernel_launch(void* const* d_in, const int* in_sizes, int n_in,
                              void* d_out, int out_size, void* d_ws, size_t ws_size,
                              hipStream_t stream) {
  (void)in_sizes; (void)n_in; (void)out_size; (void)ws_size;
  const int B = 256;
  auto F = [&](int i) { return (const float*)d_in[i]; };
  auto nb = [](size_t n) { return dim3((unsigned)((n + 255) / 256)); };

  // ---- workspace bump allocator ----
  char* wsb = (char*)d_ws; size_t off = 0;
  auto alloc = [&](size_t bytes) -> void* {
    size_t a = (bytes + 255) & ~(size_t)255; void* p = wsb + off; off += a; return p;
  };
  float* P0 = (float*)alloc((size_t)32 * 1024 * 1024 * 4);  // big mids / mlp mid
  float* P1 = (float*)alloc((size_t)8  * 1024 * 1024 * 4);  // h_spa / act ping
  float* P2 = (float*)alloc((size_t)8  * 1024 * 1024 * 4);  // n_buf / act pong
  float* P3 = (float*)alloc((size_t)12 * 1024 * 1024 * 4);  // cat / qkv out
  float* P4 = (float*)alloc((size_t)4  * 1024 * 1024 * 4);  // attn o / f1
  float* X1   = (float*)alloc((size_t)B * 30 * 512 * 4);
  float* X2   = (float*)alloc((size_t)B * 60 * 256 * 4);
  float* X3   = (float*)alloc((size_t)B * 120 * 128 * 4);
  float* X4   = (float*)alloc((size_t)B * 240 * 64 * 4);
  float* XSPA = (float*)alloc((size_t)B * 30 * 256 * 4);

  auto cvt = [&](int idx, size_t n) -> __bf16* {
    __bf16* p = (__bf16*)alloc(n * 2);
    k_f32_to_bf16<<<nb(n), 256, 0, stream>>>(F(idx), p, n);
    return p;
  };

  // ---- pytree leaf indices (sorted dict keys) ----
  // 0 ca.fc1, 1 ca.fc2, 2 conv_b, 3 conv_w,
  // down1: 4 proj_b 5 proj_w 6 qkv 7 b1 8 b2 9 g1 10 g2 11 w1 12 w2
  // down2: 13..21  down3: 22..30
  // inc: 31 b1 32 b2 33 g1 34 g2 35 w1 36 w2
  // 37 lin1_b 38 lin1_w 39 lin2_b 40 lin2_w 41 lin3_b 42 lin3_w
  // spatial blocks: 43+blk*10 + {fc1_b,fc1_w,fc2_b,fc2_w,n1b,n1g,n2b,n2g,spa_b,spa_w}
  // 103 norm_b 104 norm_g 105 phase_b 106 phase_w
  // up1: 107..117  up2: 118..128  up3: 129..139   x: 140

  __bf16* fc1w16[6]; __bf16* fc2w16[6];
  for (int blk = 0; blk < 6; ++blk) {
    int base = 43 + blk * 10;
    fc1w16[blk] = cvt(base + 1, (size_t)1024 * 256);
    fc2w16[blk] = cvt(base + 3, (size_t)256 * 1024);
  }
  __bf16* qkvD1 = cvt(6,  (size_t)768 * 256); __bf16* prjD1 = cvt(5,  (size_t)256 * 256);
  __bf16* qkvD2 = cvt(15, (size_t)384 * 128); __bf16* prjD2 = cvt(14, (size_t)128 * 128);
  __bf16* qkvD3 = cvt(24, (size_t)192 * 64);  __bf16* prjD3 = cvt(23, (size_t)64 * 64);
  __bf16* qkvU1 = cvt(109,(size_t)192 * 64);  __bf16* prjU1 = cvt(108,(size_t)64 * 64);
  __bf16* qkvU2 = cvt(120,(size_t)384 * 128); __bf16* prjU2 = cvt(119,(size_t)128 * 128);
  __bf16* qkvU3 = cvt(131,(size_t)768 * 256); __bf16* prjU3 = cvt(130,(size_t)256 * 256);
  __bf16* lin1w = cvt(38, (size_t)1280 * 7680);
  __bf16* lin2w = cvt(40, (size_t)256 * 1280);

  auto gemm = [&](const float* A, const __bf16* W, const float* bias, const float* res,
                  float* C, int M, int N, int K, int postop) {
    dim3 g((unsigned)((N + 127) / 128), (unsigned)((M + 63) / 64));
    k_gemm_wmma<<<g, 256, 0, stream>>>(A, W, bias, res, C, M, N, K, postop);
  };
  auto attn = [&](const float* xin, float* xout, const __bf16* qkvw, const __bf16* projw,
                  int projb_idx, int Ntok, int Cd) {
    int M = B * Ntok;
    gemm(xin, qkvw, nullptr, nullptr, P3, M, 3 * Cd, Cd, 0);
    int hd = Cd / 8;
    if (hd == 32)      k_attn_core<32><<<dim3(B * 8), 256, 0, stream>>>(P3, P4, Ntok, Cd);
    else if (hd == 16) k_attn_core<16><<<dim3(B * 8), 256, 0, stream>>>(P3, P4, Ntok, Cd);
    else               k_attn_core<8><<<dim3(B * 8), 256, 0, stream>>>(P3, P4, Ntok, Cd);
    gemm(P4, projw, F(projb_idx), nullptr, xout, M, Cd, Cd, 0);
  };
  auto dconv = [&](const float* xin, float* xout, int Ci, int Co, int L,
                   int w1i, int g1i, int b1i, int w2i, int g2i, int b2i) {
    int Cm = 4 * Ci;
    k_conv1d_k3<<<nb((size_t)B * Cm * L), 256, 0, stream>>>(xin, F(w1i), P0, B, Ci, Cm, L);
    k_ln_rows<<<dim3((unsigned)(B * Cm)), 256, 0, stream>>>(P0, F(g1i), F(b1i), P0, L, 1);
    k_conv1d_k3<<<nb((size_t)B * Co * L), 256, 0, stream>>>(P0, F(w2i), xout, B, Cm, Co, L);
    k_ln_rows<<<dim3((unsigned)(B * Co)), 256, 0, stream>>>(xout, F(g2i), F(b2i), xout, L, 1);
  };

  // ======================= spatial branch ==================================
  k_phase_ln_relu<<<dim3(B * 4 * 30), 256, 0, stream>>>(F(140), F(106), F(105),
                                                        F(104), F(103), P1, B, 30, 256);
  for (int blk = 0; blk < 6; ++blk) {
    int base = 43 + blk * 10;
    k_ln_rows<<<dim3(B * 4 * 30), 256, 0, stream>>>(P1, F(base + 5), F(base + 4), P2, 256, 0);
    k_mean_max_c4<<<nb((size_t)B * 7680), 256, 0, stream>>>(P2, P3, B, 7680);
    k_spa_gate<<<nb((size_t)B * 7680), 256, 0, stream>>>(P2, P3, F(base + 9), F(base + 8),
                                                         P1, B, 30, 256);
    k_ln_rows<<<dim3(B * 4 * 30), 256, 0, stream>>>(P1, F(base + 7), F(base + 6), P2, 256, 0);
    gemm(P2, fc1w16[blk], F(base + 0), nullptr, P0, 30720, 1024, 256, 1);
    gemm(P0, fc2w16[blk], F(base + 2), P1, P1, 30720, 256, 1024, 0);
  }
  k_ln_rows<<<dim3(B * 4 * 30), 256, 0, stream>>>(P1, F(104), F(103), P2, 256, 0);
  k_mean_c4<<<nb((size_t)B * 7680), 256, 0, stream>>>(P2, XSPA, B, 7680);

  // ======================= U-Net branch ====================================
  k_inc_concat<<<nb((size_t)B * 30 * 512), 256, 0, stream>>>(F(140), P2, B, 30, 256);
  dconv(P2, X1, 30, 30, 512, 35, 33, 31, 36, 34, 32);            // x1 [B,30,512]

  k_pool2<<<nb((size_t)B * 30 * 256), 256, 0, stream>>>(X1, P1, (size_t)B * 30 * 256);
  attn(P1, P1, qkvD1, prjD1, 4, 30, 256);
  dconv(P1, X2, 30, 60, 256, 11, 9, 7, 12, 10, 8);               // x2 [B,60,256]

  k_pool2<<<nb((size_t)B * 60 * 128), 256, 0, stream>>>(X2, P1, (size_t)B * 60 * 128);
  attn(P1, P1, qkvD2, prjD2, 13, 60, 128);
  dconv(P1, X3, 60, 120, 128, 20, 18, 16, 21, 19, 17);           // x3 [B,120,128]

  k_pool2<<<nb((size_t)B * 120 * 64), 256, 0, stream>>>(X3, P1, (size_t)B * 120 * 64);
  attn(P1, P1, qkvD3, prjD3, 22, 120, 64);
  dconv(P1, X4, 120, 240, 64, 29, 27, 25, 30, 28, 26);           // x4 [B,240,64]

  // up1
  attn(X4, P1, qkvU1, prjU1, 107, 240, 64);
  k_convt2<<<nb((size_t)B * 120 * 128), 256, 0, stream>>>(P1, F(117), F(116), P2, B, 240, 120, 64);
  k_concat_ch<<<nb((size_t)B * 240 * 128), 256, 0, stream>>>(X3, P2, P1, B, 120, 120, 128);
  dconv(P1, P2, 240, 120, 128, 114, 112, 110, 115, 113, 111);    // y [B,120,128]
  // up2
  attn(P2, P1, qkvU2, prjU2, 118, 120, 128);
  k_convt2<<<nb((size_t)B * 60 * 256), 256, 0, stream>>>(P1, F(128), F(127), P2, B, 120, 60, 128);
  k_concat_ch<<<nb((size_t)B * 120 * 256), 256, 0, stream>>>(X2, P2, P1, B, 60, 60, 256);
  dconv(P1, P2, 120, 60, 256, 125, 123, 121, 126, 124, 122);     // y [B,60,256]
  // up3
  attn(P2, P1, qkvU3, prjU3, 129, 60, 256);
  k_convt2<<<nb((size_t)B * 30 * 512), 256, 0, stream>>>(P1, F(139), F(138), P2, B, 60, 30, 256);
  k_concat_ch<<<nb((size_t)B * 60 * 512), 256, 0, stream>>>(X1, P2, P1, B, 30, 30, 512);
  dconv(P1, P2, 60, 30, 512, 136, 134, 132, 137, 135, 133);      // y [B,30,512]

  // ======================= head ============================================
  k_conv_s2_add<<<nb((size_t)B * 30 * 256), 256, 0, stream>>>(P2, F(3), F(2), XSPA, P1,
                                                              B, 30, 30, 512);
  k_chan_att<<<dim3(B), 256, 0, stream>>>(P1, F(0), F(1), B, 30, 256);
  gemm(P1, lin1w, F(37), nullptr, P4, 256, 1280, 7680, 1);
  gemm(P4, lin2w, F(39), nullptr, P3, 256, 256, 1280, 1);
  k_lin3<<<nb((size_t)B * 4), 256, 0, stream>>>(P3, F(42), F(41), (float*)d_out, B, 256, 4);
}
